// MutilAttention_71743133712748
// MI455X (gfx1250) — matrix operand order repaired and checked by reading, not compiled, not run on hardware
//
#include <hip/hip_runtime.h>
#include <hip/hip_bf16.h>

#define N_TOK  4096
#define IN_CH  256
#define OUT_CH 64
#define HEADS  8
#define PROJ   (OUT_CH * HEADS)   // 512
#define NLB    (N_TOK / 128)      // 32 key blocks

typedef __bf16 bf16_t;
typedef unsigned int u32;
typedef unsigned long long u64;
typedef __attribute__((ext_vector_type(16))) __bf16 v16bf;
typedef __attribute__((ext_vector_type(8)))  __bf16 v8bf;
typedef __attribute__((ext_vector_type(8)))  float  v8f;
typedef __attribute__((ext_vector_type(4)))  unsigned int v4u;
typedef __attribute__((ext_vector_type(8)))  int v8i;
typedef __attribute__((ext_vector_type(4)))  int v4i;

#define WMMA_BF16(a, b, c) \
  __builtin_amdgcn_wmma_f32_16x16x32_bf16(false, (a), false, (b), (short)0, (c), false, false)

__device__ __forceinline__ v16bf make16(v8bf lo, v8bf hh) {
  v16bf r;
#pragma unroll
  for (int i = 0; i < 8; ++i) { r[i] = lo[i]; r[i + 8] = hh[i]; }
  return r;
}

// A fragment (16x32 bf16), source tile row-major [row][k], ld multiple of 8.
__device__ __forceinline__ v16bf load_fragA(const bf16_t* t, int ld, int rowBase,
                                            int kBase, int lane) {
  const int m = lane & 15, hs = (lane >> 4) * 8;
  const bf16_t* p = t + (rowBase + m) * ld + kBase + hs;
  return make16(*(const v8bf*)p, *(const v8bf*)(p + 16));
}

// B fragment (32x16 bf16) supplied as B^T row-major [n][k], ld multiple of 8.
__device__ __forceinline__ v16bf load_fragB(const bf16_t* t, int ld, int nBase,
                                            int kBase, int lane) {
  const int n = lane & 15, ks = (lane >> 4) * 8;
  const bf16_t* p = t + (nBase + n) * ld + kBase + ks;
  return make16(*(const v8bf*)p, *(const v8bf*)(p + 16));
}

// TDM: DMA a 2D bf16 tile [tile_d1 rows x tile_d0 elems] (row stride `stride0`
// elems) from global into LDS, optional LDS row padding via padctl.
// D# packing per CDNA5 ISA (count=1, type=2,
// data_size=1 -> 2 bytes). 6-arg builtin (clang-23 / therock-10.0 signature).
__device__ __forceinline__ void tdm_load_2d(u32 lds_addr, const bf16_t* gptr,
                                            u32 tile_d0, u32 tile_d1,
                                            u32 stride0, u32 padctl) {
  const u64 ga = (u64)(uintptr_t)gptr;
  v4u g0;
  g0.x = 1u;                                   // count=1, user descriptor
  g0.y = lds_addr;                             // LDS byte address
  g0.z = (u32)ga;                              // global_addr[31:0]
  g0.w = (u32)(ga >> 32) | 0x80000000u;        // global_addr[56:32] | type=2
  v8i g1;
  g1[0] = (int)((1u << 16) | padctl);          // data_size=2B | pad controls
  g1[1] = (int)(tile_d0 << 16);                // tensor_dim0 = tile_d0
  g1[2] = (int)((tile_d0 >> 16) | (tile_d1 << 16));   // tensor_dim1 = tile_d1
  g1[3] = (int)((tile_d1 >> 16) | (tile_d0 << 16));   // tile_dim0
  g1[4] = (int)(tile_d1 & 0xffffu);            // tile_dim1 (tile_dim2 = 0)
  g1[5] = (int)stride0;                        // tensor_dim0_stride[31:0]
  g1[6] = 0;
  g1[7] = 0;
  const v4i z4 = {0, 0, 0, 0};
  const v8i z8 = {0, 0, 0, 0, 0, 0, 0, 0};
  __builtin_amdgcn_tensor_load_to_lds(g0, g1, z4, z4, z8, 0);
}

// pad_enable | pad_interval=64 DWORDs (code 5) | pad_amount=4 DWORDs (code 3):
// adds 8 bf16 of padding after each 128-elem row -> effective LDS ld = 136.
#define VT_PADCTL ((1u << 20) | (5u << 22) | (3u << 25))

__global__ void zero_kernel(float* __restrict__ p, int n) {
  int i = blockIdx.x * blockDim.x + threadIdx.x;
  if (i < n) p[i] = 0.0f;
}

// Y = X * W^T + bias via bf16 WMMA. transposed==0: Y[row][col] ([4096,512]);
// transposed!=0: Y[col][row] ([512,4096]) so attention can TDM-load V^T tiles.
__global__ __launch_bounds__(256) void proj_kernel(const float* __restrict__ X,
                                                   const float* __restrict__ W,
                                                   const float* __restrict__ bias,
                                                   bf16_t* __restrict__ Y,
                                                   int transposed) {
  __shared__ __align__(16) bf16_t Xt[128 * 64];
  __shared__ __align__(16) bf16_t Wt[128 * 64];
  const int tid  = threadIdx.x;
  const int lane = tid & 31;
  const int w    = tid >> 5;
  const int cb   = blockIdx.x;
  const int rb   = blockIdx.y;

  v8f acc[8];
#pragma unroll
  for (int i = 0; i < 8; ++i)
#pragma unroll
    for (int j = 0; j < 8; ++j) acc[i][j] = 0.0f;

  for (int kc = 0; kc < IN_CH; kc += 64) {
    for (int idx = tid; idx < 128 * 64; idx += 256) {
      const int r = idx >> 6, c = idx & 63;
      Xt[idx] = (bf16_t)X[(rb * 128 + r) * IN_CH + kc + c];
      Wt[idx] = (bf16_t)W[(cb * 128 + r) * IN_CH + kc + c];
    }
    __syncthreads();
    const v16bf a0 = load_fragA(Xt, 64, w * 16, 0, lane);
    const v16bf a1 = load_fragA(Xt, 64, w * 16, 32, lane);
    v16bf b0[8], b1[8];
#pragma unroll
    for (int nt = 0; nt < 8; ++nt) {
      b0[nt] = load_fragB(Wt, 64, nt * 16, 0, lane);
      b1[nt] = load_fragB(Wt, 64, nt * 16, 32, lane);
    }
#pragma unroll
    for (int nt = 0; nt < 8; ++nt) {
      acc[nt] = WMMA_BF16(a0, b0[nt], acc[nt]);
      acc[nt] = WMMA_BF16(a1, b1[nt], acc[nt]);
    }
    __syncthreads();
  }

  const int n = lane & 15, hi2 = lane >> 4;
#pragma unroll
  for (int nt = 0; nt < 8; ++nt) {
    const int col = cb * 128 + nt * 16 + n;
    const float bv = bias[col];
#pragma unroll
    for (int r = 0; r < 8; ++r) {
      const int row = rb * 128 + w * 16 + r + 8 * hi2;
      const bf16_t val = (bf16_t)(acc[nt][r] + bv);
      if (transposed)
        Y[(size_t)col * N_TOK + row] = val;
      else
        Y[(size_t)row * PROJ + col] = val;
    }
  }
}

// Fused sigmoid attention, one (head, 128-query-row block) per workgroup.
// K/V tiles staged by the Tensor Data Mover into double-buffered LDS.
__global__ __launch_bounds__(256) void attn_kernel(const bf16_t* __restrict__ Qw,
                                                   const bf16_t* __restrict__ Kw,
                                                   const bf16_t* __restrict__ Vtg,
                                                   float* __restrict__ out) {
  __shared__ __align__(16) bf16_t Qt[128 * 64];
  __shared__ __align__(16) bf16_t Kt[2][128 * 64];
  __shared__ __align__(16) bf16_t Vt[2][64 * 136];   // V^T tile, padded ld=136
  __shared__ __align__(16) bf16_t St[128 * 128];
  __shared__ float nrm[128];

  const int tid  = threadIdx.x;
  const int lane = tid & 31;
  const int w    = tid >> 5;
  const int mb   = blockIdx.x & 31;
  const int h    = blockIdx.x >> 5;
  const int hc   = h * OUT_CH;
  const int n    = lane & 15, hi2 = lane >> 4;
  const bool issuer = (tid < 32);   // wave 0 drives the TDM

  if (issuer) {
    // Q tile [128 rows x 64] from Qw (row stride PROJ)
    tdm_load_2d((u32)(uintptr_t)&Qt[0], Qw + (size_t)(mb * 128) * PROJ + hc,
                64, 128, PROJ, 0);
    // K tile 0 [128 x 64], V^T tile 0 [64 rows x 128] (row stride N_TOK, padded)
    tdm_load_2d((u32)(uintptr_t)&Kt[0][0], Kw + 0 * PROJ + hc, 64, 128, PROJ, 0);
    tdm_load_2d((u32)(uintptr_t)&Vt[0][0], Vtg + (size_t)hc * N_TOK + 0,
                128, 64, N_TOK, VT_PADCTL);
  }
  if (tid < 128) nrm[tid] = 0.0f;

  v8f oacc[4];
#pragma unroll
  for (int i = 0; i < 4; ++i)
#pragma unroll
    for (int j = 0; j < 8; ++j) oacc[i][j] = 0.0f;

  for (int ib = 0; ib < NLB; ++ib) {
    const int cur = ib & 1;
    if (issuer) __builtin_amdgcn_s_wait_tensorcnt(0);
    __syncthreads();   // staged tiles visible; previous iteration's reads done

    if (issuer && (ib + 1 < NLB)) {
      const int lb1 = (ib + 1) * 128;
      tdm_load_2d((u32)(uintptr_t)&Kt[cur ^ 1][0], Kw + (size_t)lb1 * PROJ + hc,
                  64, 128, PROJ, 0);
      tdm_load_2d((u32)(uintptr_t)&Vt[cur ^ 1][0], Vtg + (size_t)hc * N_TOK + lb1,
                  128, 64, N_TOK, VT_PADCTL);
    }

    // ---- S = sigmoid(Q K^T): this wave's 16-row strip x 128 key cols ----
    const v16bf a0 = load_fragA(Qt, 64, w * 16, 0, lane);
    const v16bf a1 = load_fragA(Qt, 64, w * 16, 32, lane);
    v16bf b0[8], b1[8];
#pragma unroll
    for (int nt = 0; nt < 8; ++nt) {
      b0[nt] = load_fragB(Kt[cur], 64, nt * 16, 0, lane);
      b1[nt] = load_fragB(Kt[cur], 64, nt * 16, 32, lane);
    }
    float rsum[8];
#pragma unroll
    for (int r = 0; r < 8; ++r) rsum[r] = 0.0f;
#pragma unroll
    for (int nt = 0; nt < 8; ++nt) {
      v8f s;
#pragma unroll
      for (int j = 0; j < 8; ++j) s[j] = 0.0f;
      s = WMMA_BF16(a0, b0[nt], s);
      s = WMMA_BF16(a1, b1[nt], s);
#pragma unroll
      for (int r = 0; r < 8; ++r) {
        const float sig = 1.0f / (1.0f + __expf(-s[r]));
        rsum[r] += sig;
        St[(w * 16 + r + 8 * hi2) * 128 + nt * 16 + n] = (bf16_t)sig;
      }
    }
#pragma unroll
    for (int r = 0; r < 8; ++r)
      atomicAdd(&nrm[w * 16 + r + 8 * hi2], rsum[r]);   // ds_add_f32
    __syncthreads();   // St complete

    // ---- O += S * V (A = St, B^T = Vt) ----
    v16bf sa[4];
    v16bf vb[4][4];
#pragma unroll
    for (int kk = 0; kk < 4; ++kk) {
      sa[kk] = load_fragA(St, 128, w * 16, kk * 32, lane);
#pragma unroll
      for (int dt = 0; dt < 4; ++dt)
        vb[kk][dt] = load_fragB(Vt[cur], 136, dt * 16, kk * 32, lane);
    }
#pragma unroll
    for (int kk = 0; kk < 4; ++kk)
#pragma unroll
      for (int dt = 0; dt < 4; ++dt)
        oacc[dt] = WMMA_BF16(sa[kk], vb[kk][dt], oacc[dt]);
    // next iteration's top barrier protects St / Kt / Vt reuse
  }

  // out[n,d] += (1/HEADS) * O[row,d] / nrm[row]
#pragma unroll
  for (int r = 0; r < 8; ++r) {
    const int row = w * 16 + r + 8 * hi2;
    const float inv = 0.125f / nrm[row];
#pragma unroll
    for (int dt = 0; dt < 4; ++dt) {
      const float v = oacc[dt][r] * inv;
      __hip_atomic_fetch_add(&out[(mb * 128 + row) * OUT_CH + dt * 16 + n], v,
                             __ATOMIC_RELAXED, __HIP_MEMORY_SCOPE_AGENT);
    }
  }
}

extern "C" void kernel_launch(void* const* d_in, const int* in_sizes, int n_in,
                              void* d_out, int out_size, void* d_ws, size_t ws_size,
                              hipStream_t stream) {
  const float* X  = (const float*)d_in[0];
  const float* Wq = (const float*)d_in[1];
  const float* bq = (const float*)d_in[2];
  const float* Wk = (const float*)d_in[3];
  const float* bk = (const float*)d_in[4];
  const float* Wv = (const float*)d_in[5];
  const float* bv = (const float*)d_in[6];
  float* out = (float*)d_out;

  bf16_t* Qw  = (bf16_t*)d_ws;
  bf16_t* Kw  = Qw + (size_t)N_TOK * PROJ;
  bf16_t* Vtg = Kw + (size_t)N_TOK * PROJ;   // V projection stored transposed

  zero_kernel<<<(N_TOK * OUT_CH + 255) / 256, 256, 0, stream>>>(out, N_TOK * OUT_CH);

  dim3 pg(PROJ / 128, N_TOK / 128);
  proj_kernel<<<pg, 256, 0, stream>>>(X, Wq, bq, Qw, 0);
  proj_kernel<<<pg, 256, 0, stream>>>(X, Wk, bk, Kw, 0);
  proj_kernel<<<pg, 256, 0, stream>>>(X, Wv, bv, Vtg, 1);

  attn_kernel<<<dim3((N_TOK / 128) * HEADS), 256, 0, stream>>>(Qw, Kw, Vtg, out);
}
